// ColBERT_7816840478780
// MI455X (gfx1250) — matrix operand order repaired and checked by reading, compiled, not run on hardware
//
#include <hip/hip_runtime.h>
#include <math.h>

typedef __bf16 bf16_t;
typedef __attribute__((ext_vector_type(16))) __bf16 bf16x16;
typedef __attribute__((ext_vector_type(8)))  float  f32x8;

union FragU { bf16x16 v; uint4 u[2]; };

// ---- CDNA5 async global->LDS copy (ASYNCcnt-tracked), 16B per lane ----
__device__ inline void async_copy_b128(unsigned lds_addr, const void* gptr) {
    asm volatile("global_load_async_to_lds_b128 %0, %1, off"
                 :: "v"(lds_addr), "v"((unsigned long long)(uintptr_t)gptr)
                 : "memory");
}
__device__ inline void wait_async0() {
    asm volatile("s_wait_asynccnt 0" ::: "memory");
}

// A-matrix 16x32 bf16 fragment from row-major bf16 [rows x stride].
// Lane holds row (lane&15); VGPR0-3: K in [half*8, half*8+8), VGPR4-7: K in [16+half*8, ...).
__device__ inline bf16x16 load_fragA_bf16(const bf16_t* __restrict__ base, int stride,
                                          int row, int kk, int lane) {
    int lh = lane >> 4;
    const bf16_t* p = base + (size_t)row * stride + kk + lh * 8;
    FragU f;
    f.u[0] = *(const uint4*)(p);
    f.u[1] = *(const uint4*)(p + 16);
    return f.v;
}

// B-matrix 32x16 bf16 fragment: lane holds column n (rowN in B^T storage),
// K in [half*8, half*8+8) then [16+half*8, 16+half*8+8).
__device__ inline bf16x16 load_fragB_bf16(const bf16_t* __restrict__ base, int stride,
                                          int rowN, int kk, int lane) {
    int lh = lane >> 4;
    const bf16_t* p = base + (size_t)rowN * stride + kk + lh * 8;
    FragU f;
    f.u[0] = *(const uint4*)(p);
    f.u[1] = *(const uint4*)(p + 16);
    return f.v;
}

// B fragment from an LDS-resident 16x128 bf16 tile.
__device__ inline bf16x16 load_fragB_lds(const bf16_t* base, int kk, int lane) {
    int lh = lane >> 4;
    const bf16_t* p = base + (size_t)(lane & 15) * 128 + kk + lh * 8;
    FragU f;
    f.u[0] = *(const uint4*)(p);
    f.u[1] = *(const uint4*)(p + 16);
    return f.v;
}

// A fragment built from f32 source row (converted to bf16 on the fly).
__device__ inline bf16x16 load_fragA_f32(const float* __restrict__ rowptr, int kk, int lane) {
    int lh = lane >> 4;
    const float4* p0 = (const float4*)(rowptr + kk + lh * 8);
    const float4* p1 = (const float4*)(rowptr + kk + 16 + lh * 8);
    float4 a0 = p0[0], a1 = p0[1];
    float4 a2 = p1[0], a3 = p1[1];
    union { bf16x16 v; bf16_t e[16]; } f;
    f.e[0]=(bf16_t)a0.x;  f.e[1]=(bf16_t)a0.y;  f.e[2]=(bf16_t)a0.z;  f.e[3]=(bf16_t)a0.w;
    f.e[4]=(bf16_t)a1.x;  f.e[5]=(bf16_t)a1.y;  f.e[6]=(bf16_t)a1.z;  f.e[7]=(bf16_t)a1.w;
    f.e[8]=(bf16_t)a2.x;  f.e[9]=(bf16_t)a2.y;  f.e[10]=(bf16_t)a2.z; f.e[11]=(bf16_t)a2.w;
    f.e[12]=(bf16_t)a3.x; f.e[13]=(bf16_t)a3.y; f.e[14]=(bf16_t)a3.z; f.e[15]=(bf16_t)a3.w;
    return f.v;
}

__global__ void k_convw(const float* __restrict__ w, bf16_t* __restrict__ wbf, int n) {
    int i = blockIdx.x * blockDim.x + threadIdx.x;
    if (i < n) wbf[i] = (bf16_t)w[i];
}

__global__ void k_qlen(const int* __restrict__ mask, float* __restrict__ qlen) {
    __shared__ int sred[128];
    int b = blockIdx.x, t = threadIdx.x;
    int v = (t >= 1) ? mask[b * 128 + t] : 0;
    sred[t] = v;
    __syncthreads();
    for (int s = 64; s > 0; s >>= 1) {
        if (t < s) sred[t] += sred[t + s];
        __syncthreads();
    }
    if (t == 0) qlen[b] = (float)sred[0];
}

// Projection + token mask + L2 normalize. One wave per 16-token output tile.
// hidden: f32 [Bn x L x 768]; output: bf16 [Bn x Lpad x 128] (padded rows zero).
__global__ void __launch_bounds__(256)
k_proj(const float* __restrict__ hidden, const int* __restrict__ ids,
       bf16_t* __restrict__ oute, int L, int Ltok, int Lpad,
       const bf16_t* __restrict__ wbf, const float* __restrict__ bias) {
    const int H = 768, C = 128;
    int wave = (blockIdx.x * blockDim.x + threadIdx.x) >> 5;
    int lane = threadIdx.x & 31;
    int lh = lane >> 4, col = lane & 15;
    int tilesPerBatch = Lpad >> 4;
    int b = wave / tilesPerBatch;
    int tile = wave % tilesPerBatch;

    // hide the post-loop token-id fetch behind the K loop
    __builtin_prefetch(ids + b * L + tile * 16, 0, 1);

    f32x8 acc[8];
#pragma unroll
    for (int ct = 0; ct < 8; ++ct) {
        float bv = bias[ct * 16 + col];
#pragma unroll
        for (int r = 0; r < 8; ++r) acc[ct][r] = bv;
    }

    int arow = tile * 16 + col;                    // output token index for A row
    int hr = arow + 1; if (hr > L - 1) hr = L - 1; // clamp (padded rows zeroed later)
    const float* rowptr = hidden + ((size_t)b * L + hr) * H;

    for (int kk = 0; kk < H; kk += 32) {
        bf16x16 a = load_fragA_f32(rowptr, kk, lane);
        // preload all 8 W fragments, then an uninterrupted burst of 8
        // independent WMMAs (distinct accumulators -> no RAW chain)
        bf16x16 bfr[8];
#pragma unroll
        for (int ct = 0; ct < 8; ++ct)
            bfr[ct] = load_fragB_bf16(wbf, H, ct * 16 + col, kk, lane);
#pragma unroll
        for (int ct = 0; ct < 8; ++ct)
            acc[ct] = __builtin_amdgcn_wmma_f32_16x16x32_bf16(
                false, a, false, bfr[ct], (short)0, acc[ct], false, false);
    }

    // token keep mask (PAD=0 / CLS=101 / out-of-range -> 0)
    float kf[8];
#pragma unroll
    for (int r = 0; r < 8; ++r) {
        int m = tile * 16 + r + lh * 8;
        float k = 0.f;
        if (m < Ltok) {
            int id = ids[b * L + m + 1];
            k = (id == 0 || id == 101) ? 0.f : 1.f;
        }
        kf[r] = k;
    }
#pragma unroll
    for (int ct = 0; ct < 8; ++ct)
#pragma unroll
        for (int r = 0; r < 8; ++r) acc[ct][r] *= kf[r];

    // L2 norm per token: sum squares over 8 column tiles, reduce across 16-lane half
    float s[8];
#pragma unroll
    for (int r = 0; r < 8; ++r) {
        float t = 0.f;
#pragma unroll
        for (int ct = 0; ct < 8; ++ct) t += acc[ct][r] * acc[ct][r];
        s[r] = t;
    }
#pragma unroll
    for (int msk = 1; msk <= 8; msk <<= 1)
#pragma unroll
        for (int r = 0; r < 8; ++r) s[r] += __shfl_xor(s[r], msk, 32);
#pragma unroll
    for (int r = 0; r < 8; ++r) {
        float inv = 1.f / fmaxf(sqrtf(s[r]), 1e-12f);
#pragma unroll
        for (int ct = 0; ct < 8; ++ct) acc[ct][r] *= inv;
    }

    // store bf16 [b][Lpad][128]
#pragma unroll
    for (int r = 0; r < 8; ++r) {
        int m = tile * 16 + r + lh * 8;
        bf16_t* orow = oute + ((size_t)b * Lpad + m) * C + col;
#pragma unroll
        for (int ct = 0; ct < 8; ++ct) orow[ct * 16] = (bf16_t)acc[ct][r];
    }
}

// MaxSim scoring: one block per (query, doc) pair; wave w handles q rows [16w,16w+16).
// Doc tiles (16x128 bf16 = 4KB) staged into LDS via async global->LDS DMA,
// double-buffered so the async copy of tile dt+1 overlaps the WMMA chain on tile dt.
// mode 0: pos (pair p -> b=p, doc p of pe, scores col 0, also writes pos_out[b]).
// mode 1: neg (pair p -> b=p>>5, n=p&31, scores col 1+n).
__global__ void __launch_bounds__(256)
k_score(const bf16_t* __restrict__ qe, const bf16_t* __restrict__ de,
        float* __restrict__ scores, float* __restrict__ pos_out,
        const float* __restrict__ qlen, int mode) {
    const int C = 128, LQP = 128, LDP = 512, LQT = 127;
    __shared__ __align__(16) bf16_t sdoc[2][16 * 128];
    __shared__ float ssum[8];
    int p = blockIdx.x;
    int b, colOut;
    const bf16_t* doc;
    if (mode == 0) { b = p; doc = de + (size_t)p * LDP * C; colOut = 0; }
    else           { b = p >> 5; int n = p & 31; doc = de + (size_t)n * LDP * C; colOut = 1 + n; }

    int tid = threadIdx.x;
    int wave = tid >> 5, lane = tid & 31;
    int lh = lane >> 4;
    const bf16_t* qb = qe + (size_t)b * LQP * C;

    bf16x16 afr[4];
#pragma unroll
    for (int k4 = 0; k4 < 4; ++k4)
        afr[k4] = load_fragA_bf16(qb, C, wave * 16 + (lane & 15), k4 * 32, lane);

    float rm[8];
#pragma unroll
    for (int r = 0; r < 8; ++r) rm[r] = -1e30f;

    // Stage tile 0: 256 threads x one async b128 each = full 4KB tile.
    {
        unsigned l0 = (unsigned)(uintptr_t)(&sdoc[0][tid * 8]);
        async_copy_b128(l0, doc + (size_t)tid * 8);
        wait_async0();
    }
    __syncthreads();

    for (int dt = 0; dt < 32; ++dt) {
        int cur = dt & 1;
        if (dt + 1 < 32) {   // kick async DMA of next tile into the other buffer
            unsigned ln = (unsigned)(uintptr_t)(&sdoc[cur ^ 1][tid * 8]);
            async_copy_b128(ln, doc + (size_t)(dt + 1) * 16 * C + tid * 8);
        }

        // preload 4 LDS B fragments, then WMMA burst with two
        // independent accumulator chains (depth 2 instead of 4)
        bf16x16 bfr[4];
#pragma unroll
        for (int k4 = 0; k4 < 4; ++k4)
            bfr[k4] = load_fragB_lds(&sdoc[cur][0], k4 * 32, lane);

        f32x8 acc0 = {0.f, 0.f, 0.f, 0.f, 0.f, 0.f, 0.f, 0.f};
        f32x8 acc1 = {0.f, 0.f, 0.f, 0.f, 0.f, 0.f, 0.f, 0.f};
        acc0 = __builtin_amdgcn_wmma_f32_16x16x32_bf16(
            false, afr[0], false, bfr[0], (short)0, acc0, false, false);
        acc1 = __builtin_amdgcn_wmma_f32_16x16x32_bf16(
            false, afr[1], false, bfr[1], (short)0, acc1, false, false);
        acc0 = __builtin_amdgcn_wmma_f32_16x16x32_bf16(
            false, afr[2], false, bfr[2], (short)0, acc0, false, false);
        acc1 = __builtin_amdgcn_wmma_f32_16x16x32_bf16(
            false, afr[3], false, bfr[3], (short)0, acc1, false, false);

        bool lastpad = (dt == 31) && ((lane & 15) == 15);  // exclude padded doc col 511
#pragma unroll
        for (int r = 0; r < 8; ++r) {
            float v = lastpad ? -1e30f : (acc0[r] + acc1[r]);
            rm[r] = fmaxf(rm[r], v);
        }

        wait_async0();       // own async done; barrier ensures tile visible block-wide
        __syncthreads();
    }

#pragma unroll
    for (int msk = 1; msk <= 8; msk <<= 1)
#pragma unroll
        for (int r = 0; r < 8; ++r) rm[r] = fmaxf(rm[r], __shfl_xor(rm[r], msk, 32));

    float part = 0.f;
#pragma unroll
    for (int r = 0; r < 8; ++r) {
        int m = wave * 16 + r + lh * 8;
        if (m < LQT) part += rm[r];   // exclude padded query row 127
    }
    part += __shfl_xor(part, 16, 32);
    if (lane == 0) ssum[wave] = part;
    __syncthreads();
    if (tid == 0) {
        float tot = 0.f;
        for (int w = 0; w < 8; ++w) tot += ssum[w];
        float sc = tot / qlen[b];
        scores[b * 33 + colOut] = sc;
        if (mode == 0) pos_out[b] = sc;
    }
}

__global__ void k_loss(const float* __restrict__ scores, float* __restrict__ out_loss) {
    int lane = threadIdx.x;  // 32 lanes, one per batch row
    const float invT = 50.0f;
    float logits[33];
    float lmax = -1e30f;
    for (int j = 0; j < 33; ++j) {
        float v = scores[lane * 33 + j] * invT;
        logits[j] = v;
        lmax = fmaxf(lmax, v);
    }
    float se = 0.f;
    for (int j = 0; j < 33; ++j) se += expf(logits[j] - lmax);
    float lb = (lmax + logf(se)) - logits[0];
    for (int m = 1; m < 32; m <<= 1) lb += __shfl_xor(lb, m, 32);
    if (lane == 0) out_loss[0] = lb / 32.0f;
}

extern "C" void kernel_launch(void* const* d_in, const int* in_sizes, int n_in,
                              void* d_out, int out_size, void* d_ws, size_t ws_size,
                              hipStream_t stream) {
    const float* q_hidden = (const float*)d_in[0];
    const int*   q_ids    = (const int*)d_in[1];
    const int*   q_mask   = (const int*)d_in[2];
    const float* p_hidden = (const float*)d_in[3];
    const int*   p_ids    = (const int*)d_in[4];
    const float* n_hidden = (const float*)d_in[6];
    const int*   n_ids    = (const int*)d_in[7];
    const float* lw       = (const float*)d_in[9];
    const float* lb       = (const float*)d_in[10];
    float* out = (float*)d_out;   // [0..31] pos_score, [32] loss

    char* ws = (char*)d_ws;
    bf16_t* wbf    = (bf16_t*)(ws);                 // 128*768*2      = 196608
    bf16_t* qe     = (bf16_t*)(ws + 196608);        // 32*128*128*2   = 1048576
    bf16_t* pe     = (bf16_t*)(ws + 1245184);       // 32*512*128*2   = 4194304
    bf16_t* ne     = (bf16_t*)(ws + 5439488);       // 32*512*128*2   = 4194304
    float*  scores = (float*)(ws + 9633792);        // 32*33*4        = 4224
    float*  qlen   = (float*)(ws + 9638016);        // 32*4

    k_convw<<<384, 256, 0, stream>>>(lw, wbf, 128 * 768);
    k_qlen<<<32, 128, 0, stream>>>(q_mask, qlen);

    // projections: waves = Bn * Lpad/16 ; 8 waves per 256-thread block
    k_proj<<<32, 256, 0, stream>>>(q_hidden, q_ids, qe, 128, 127, 128, wbf, lb);
    k_proj<<<128, 256, 0, stream>>>(p_hidden, p_ids, pe, 512, 511, 512, wbf, lb);
    k_proj<<<128, 256, 0, stream>>>(n_hidden, n_ids, ne, 512, 511, 512, wbf, lb);

    k_score<<<32, 256, 0, stream>>>(qe, pe, scores, out, qlen, 0);
    k_score<<<1024, 256, 0, stream>>>(qe, ne, scores, out, qlen, 1);

    k_loss<<<1, 32, 0, stream>>>(scores, out + 32);
}
